// IAM_77653008712342
// MI455X (gfx1250) — matrix operand order repaired and checked by reading, not compiled, not run on hardware
//
#include <hip/hip_runtime.h>
#include <hip/hip_bf16.h>

typedef __attribute__((ext_vector_type(16))) _Float16 v16h;
typedef __attribute__((ext_vector_type(8)))  float    v8f;

// Fragment <-> 16B-chunk view (v16h = 32 bytes = 2 x float4)
union F16frag { v16h h; float4 f4[2]; };

// Problem constants: features [4,256,64,64], CI=32, N=HW=4096
#define BATCH 4
#define CCH   256
#define CI    32
#define NPIX  4096
#define OUT_ELEMS ((size_t)BATCH * CCH * NPIX)           // 4,194,304 floats
#define SIM_ELEMS ((size_t)BATCH * NPIX * NPIX)          // 67,108,864 floats

// ---------------- Kernel 0: pack weights to f16 + bias array ----------------
__global__ __launch_bounds__(256) void IAM_pack_w(
    const float* __restrict__ w_red, const float* __restrict__ b_red,
    const float* __restrict__ w_val, const float* __restrict__ b_val,
    _Float16* __restrict__ wh, float* __restrict__ biasArr)
{
    int i = blockIdx.x * 256 + threadIdx.x;              // 288*256 total
    if (i < 288 * 256) {
        int o = i >> 8, c = i & 255;
        float v = (o < 32) ? w_red[o * 256 + c] : w_val[(o - 32) * 256 + c];
        wh[i] = (_Float16)v;
    }
    if (i < 288) biasArr[i] = (i < 32) ? b_red[i] : b_val[i - 32];
}

// ---------------- Kernel 1: prep GEMM: x[64x256] * whT[256x288] -> Ah, Vh ----
__global__ __launch_bounds__(256) void IAM_prep(
    const float* __restrict__ feat, const _Float16* __restrict__ wh,
    const float* __restrict__ biasArr,
    _Float16* __restrict__ Ah, _Float16* __restrict__ Vh)
{
    __shared__ __align__(16) _Float16 xh[64 * 264];      // [pix][c], pad 8

    const int tid  = threadIdx.x;
    const int lane = tid & 31;
    const int wv   = tid >> 5;
    const int hf   = lane >> 4;
    const int ln16 = lane & 15;

    const int bIdx = blockIdx.x >> 6;
    const int tile = blockIdx.x & 63;
    const float* fB = feat + (size_t)bIdx * CCH * NPIX + tile * 64;

    // stage x tile (f32 -> f16), coalesced global reads
    #pragma unroll 4
    for (int j = 0; j < 64; ++j) {
        int flat = j * 256 + tid;                        // 16384
        int c = flat >> 6, px = flat & 63;
        xh[px * 264 + c] = (_Float16)fB[(size_t)c * NPIX + px];
    }
    __syncthreads();

    // 4 m-tiles x 18 o-tiles = 72 output tiles; 9 per wave
    for (int k = 0; k < 9; ++k) {
        int idx = wv + 8 * k;                            // 0..71
        int mt = idx / 18;
        int ot = idx % 18;
        int arow = mt * 16 + ln16;
        int orow = ot * 16 + ln16;
        v8f acc = {};
        #pragma unroll
        for (int ks = 0; ks < 8; ++ks) {
            F16frag a, b;
            a.f4[0] = *(const float4*)&xh[arow * 264 + ks * 32 + hf * 16];
            a.f4[1] = *(const float4*)&xh[arow * 264 + ks * 32 + 8 + hf * 16];
            const float4* pb = (const float4*)(wh + orow * 256 + ks * 32 + hf * 16);
            b.f4[0] = pb[0]; b.f4[1] = pb[1];
            acc = __builtin_amdgcn_wmma_f32_16x16x32_f16(
                false, a.h, false, b.h, (short)0, acc, false, false);
        }
        float bias = biasArr[ot * 16 + ln16];
        int pixBase = tile * 64 + mt * 16;
        #pragma unroll
        for (int r = 0; r < 8; ++r) {
            int px = pixBase + hf * 8 + r;
            float v = acc[r] + bias;
            v = v > 0.0f ? v : 0.0f;
            size_t nG = (size_t)bIdx * NPIX + px;
            if (ot < 2) Ah[nG * CI  + ot * 16 + ln16]        = (_Float16)v;
            else        Vh[nG * CCH + (ot - 2) * 16 + ln16]  = (_Float16)v;
        }
    }
}

// ---------------- Kernel 2: sq[n] = sum(Ah[n,:]^2) ---------------------------
__global__ __launch_bounds__(256) void IAM_sq(
    const _Float16* __restrict__ Ah, float* __restrict__ sq)
{
    int n = blockIdx.x * 256 + threadIdx.x;              // 16384
    const _Float16* p = Ah + (size_t)n * CI;
    float s = 0.0f;
    #pragma unroll
    for (int k = 0; k < CI; ++k) { float v = (float)p[k]; s += v * v; }
    sq[n] = s;
}

// ---------------- Kernel 3: fused sim + GEMM2 + residual ---------------------
// Block: 64 m-rows of one batch. Per 64-n chunk: sim tile via WMMA, write sim
// (f32 out NT-streamed + f16 LDS), then acc[64x256] += sim * V via WMMA.
// Epilogue: transpose acc through LDS, add features, coalesced NT store.
__global__ __launch_bounds__(256) void IAM_fused(
    const float* __restrict__ feat, const _Float16* __restrict__ Ah,
    const float* __restrict__ sq,  const _Float16* __restrict__ Vh,
    float* __restrict__ outp)
{
    constexpr int SIM_LD = 72;                            // halves per sSim row
    constexpr int SV_LD  = 72;                            // halves per sVT row
    __shared__ __align__(16) union {
        struct { _Float16 sSim[64 * SIM_LD]; _Float16 sVT[256 * SV_LD]; } p1;
        float sAccT[128 * 65];
    } sm;

    const int tid  = threadIdx.x;
    const int lane = tid & 31;
    const int wv   = tid >> 5;
    const int hf   = lane >> 4;
    const int ln16 = lane & 15;

    const int bIdx   = blockIdx.x >> 6;
    const int mBlock = (blockIdx.x & 63) * 64;

    const _Float16* AhB = Ah + (size_t)bIdx * NPIX * CI;
    const _Float16* VhB = Vh + (size_t)bIdx * NPIX * CCH;
    const float*    sqB = sq + (size_t)bIdx * NPIX;
    float* simOut = outp + OUT_ELEMS + (size_t)bIdx * NPIX * NPIX;

    // Resident A_m fragments + sq_m: this wave's two sim m-tiles (mt0, mt0+2)
    const int mt0 = wv >> 2;                              // 0 or 1
    const int ntW = wv & 3;
    F16frag am[2];
    float sqm_[2][8];
    #pragma unroll
    for (int j = 0; j < 2; ++j) {
        int mt = mt0 + j * 2;
        int row = mBlock + mt * 16 + ln16;
        am[j].f4[0] = *(const float4*)(AhB + row * CI + hf * 16);
        am[j].f4[1] = *(const float4*)(AhB + row * CI + 8 + hf * 16);
        #pragma unroll
        for (int r = 0; r < 8; ++r)
            sqm_[j][r] = sqB[mBlock + mt * 16 + hf * 8 + r];
    }

    // Phase-B output strip for this wave: [16m x 128c]
    const int miB = wv >> 1;
    const int ciB = wv & 1;
    v8f acc[8] = {};

    for (int n0 = 0; n0 < NPIX; n0 += 64) {
        // ---- stage V chunk transposed: sVT[c][n], coalesced dword reads ----
        const uint32_t* src = (const uint32_t*)(VhB + (size_t)n0 * CCH); // 8192 dw
        #pragma unroll 4
        for (int j = 0; j < 32; ++j) {
            int dw = j * 256 + tid;
            int n = dw >> 7, c2 = dw & 127;
            union { uint32_t u; _Float16 h[2]; } cv; cv.u = src[dw];
            sm.p1.sVT[(2 * c2)     * SV_LD + n] = cv.h[0];
            sm.p1.sVT[(2 * c2 + 1) * SV_LD + n] = cv.h[1];
        }
        // ---- phase A: two 16x16 sim tiles per wave --------------------------
        #pragma unroll
        for (int j = 0; j < 2; ++j) {
            int mt = mt0 + j * 2;
            int nrow = n0 + ntW * 16 + ln16;
            F16frag bn;
            const float4* p = (const float4*)(AhB + nrow * CI + hf * 16);
            bn.f4[0] = p[0]; bn.f4[1] = p[1];
            v8f d = {};
            d = __builtin_amdgcn_wmma_f32_16x16x32_f16(
                false, am[j].h, false, bn.h, (short)0, d, false, false);
            float sqn = sqB[nrow];
            #pragma unroll
            for (int r = 0; r < 8; ++r) {
                int mL = mt * 16 + hf * 8 + r;
                float D = sqm_[j][r] + sqn - 2.0f * d[r];
                float s = 1.0f / (1.0f + __expf(-__expf(-D)));
                // 268 MB write-once stream: bypass L2 residency (TH=NT)
                __builtin_nontemporal_store(
                    s, &simOut[(size_t)(mBlock + mL) * NPIX + nrow]);
                sm.p1.sSim[mL * SIM_LD + ntW * 16 + ln16] = (_Float16)s;
            }
        }
        __syncthreads();
        // ---- phase B: acc[16x128] += sim[16x64] * V[64x128] -----------------
        #pragma unroll
        for (int kh = 0; kh < 2; ++kh) {
            F16frag a;
            int row = miB * 16 + ln16;
            a.f4[0] = *(const float4*)&sm.p1.sSim[row * SIM_LD + kh * 32 + hf * 16];
            a.f4[1] = *(const float4*)&sm.p1.sSim[row * SIM_LD + kh * 32 + 8 + hf * 16];
            #pragma unroll
            for (int ct = 0; ct < 8; ++ct) {
                F16frag b;
                int c = ciB * 128 + ct * 16 + ln16;
                const float4* pb = (const float4*)&sm.p1.sVT[c * SV_LD + kh * 32 + hf * 16];
                b.f4[0] = pb[0]; b.f4[1] = pb[1];
                acc[ct] = __builtin_amdgcn_wmma_f32_16x16x32_f16(
                    false, a.h, false, b.h, (short)0, acc[ct], false, false);
            }
        }
        __syncthreads();
    }

    // ---- epilogue: transpose acc via LDS, add residual, coalesced NT store --
    for (int cb = 0; cb < 2; ++cb) {
        __syncthreads();
        if ((wv & 1) == cb) {
            #pragma unroll
            for (int ct = 0; ct < 8; ++ct) {
                int cl = ct * 16 + ln16;                  // 0..127
                #pragma unroll
                for (int r = 0; r < 8; ++r)
                    sm.sAccT[cl * 65 + miB * 16 + hf * 8 + r] = acc[ct][r];
            }
        }
        __syncthreads();
        #pragma unroll 4
        for (int j = 0; j < 32; ++j) {
            int flat = j * 256 + tid;                     // 8192
            int m = flat & 63, cl = flat >> 6;
            int c = cb * 128 + cl;
            size_t gi = ((size_t)bIdx * CCH + c) * NPIX + mBlock + m;
            float f = __builtin_nontemporal_load(&feat[gi]);   // read-once
            __builtin_nontemporal_store(sm.sAccT[cl * 65 + m] + f, &outp[gi]);
        }
    }
}

// ---------------- Host launcher ---------------------------------------------
extern "C" void kernel_launch(void* const* d_in, const int* in_sizes, int n_in,
                              void* d_out, int out_size, void* d_ws, size_t ws_size,
                              hipStream_t stream) {
    (void)in_sizes; (void)n_in; (void)out_size; (void)ws_size;
    const float* feat  = (const float*)d_in[0];
    const float* w_red = (const float*)d_in[1];
    const float* b_red = (const float*)d_in[2];
    const float* w_val = (const float*)d_in[3];
    const float* b_val = (const float*)d_in[4];
    float* outp = (float*)d_out;

    char* ws = (char*)d_ws;                               // ~9.66 MB used
    _Float16* wh      = (_Float16*)(ws + 0);              // 288*256*2 = 147456
    float*    biasArr = (float*)   (ws + 147456);         // 288*4     = 1152
    _Float16* Ah      = (_Float16*)(ws + 148608);         // 4*4096*32*2 = 1 MiB
    float*    sqArr   = (float*)   (ws + 1197184);        // 4*4096*4  = 64 KiB
    _Float16* Vh      = (_Float16*)(ws + 1262720);        // 4*4096*256*2 = 8 MiB

    IAM_pack_w<<<288, 256, 0, stream>>>(w_red, b_red, w_val, b_val, wh, biasArr);
    IAM_prep  <<<BATCH * 64, 256, 0, stream>>>(feat, wh, biasArr, Ah, Vh);
    IAM_sq    <<<BATCH * NPIX / 256, 256, 0, stream>>>(Ah, sqArr);
    IAM_fused <<<BATCH * 64, 256, 0, stream>>>(feat, Ah, sqArr, Vh, outp);
}
